// DecoderLayer_1606317768819
// MI455X (gfx1250) — matrix operand order repaired and checked by reading, not compiled, not run on hardware
//
#include <hip/hip_runtime.h>
#include <hip/hip_bf16.h>
#include <math.h>

typedef __bf16 bf16_t;
typedef __attribute__((ext_vector_type(16))) __bf16 v16bf;
typedef __attribute__((ext_vector_type(8)))  float  v8f;
typedef __attribute__((ext_vector_type(4)))  unsigned v4u;
typedef __attribute__((ext_vector_type(8)))  int      v8i;
typedef __attribute__((ext_vector_type(4)))  int      v4i;

#if __has_builtin(__builtin_amdgcn_tensor_load_to_lds) && __has_builtin(__builtin_amdgcn_s_wait_tensorcnt)
#define HAVE_TDM 1
#else
#define HAVE_TDM 0
#endif

static constexpr int Bn = 8, Ln = 1024, Sn = 2048, Dn = 512, Hn = 8, DKn = 64, DVn = 64, HIDn = 2048;

__device__ __forceinline__ bf16_t tobf(float f) { return (bf16_t)f; }

// ---------------------------------------------------------------------------
// Generic WMMA GEMM:  C[M,N] = act(A[M,K] @ W[K,N] + bias[N])
//   A: fp32 or bf16 (ABF), row-major.  W: fp32 row-major (bf16-converted in LDS).
//   outF (fp32) and/or outB (bf16) may be null.
//   Block: 256 threads = 8 waves; tile 128x128, BK=32; wave -> 32x64 (2x4 WMMA).
//   bf16-A tiles are staged by the Tensor Data Mover (padded-LDS DMA) when
//   available; W tiles are converted f32->bf16 by the waves in parallel.
// ---------------------------------------------------------------------------
template<int ACT, bool ABF>
__global__ __launch_bounds__(256)
void gemm_wmma(const void* __restrict__ Ap, const float* __restrict__ W,
               const float* __restrict__ bias, float* __restrict__ outF,
               bf16_t* __restrict__ outB, int M, int N, int K)
{
  __shared__ __align__(16) bf16_t As[128][40];  // 128 rows x 32 k (+8 pad = 80B stride)
  __shared__ __align__(16) bf16_t Bs[128][40];  // 128 cols x 32 k (W transposed, +pad)
  const int tid  = threadIdx.x;
  const int lane = tid & 31, wave = tid >> 5;
  const int wm = wave >> 1, wn = wave & 1;      // 4 x 2 wave grid
  const int rowBase = blockIdx.y * 128;
  const int colBase = blockIdx.x * 128;
  const int ln15 = lane & 15;

  v8f acc[2][4] = {};
  const float*  Af = (const float*)Ap;
  const bf16_t* Ab = (const bf16_t*)Ap;

  for (int kt = 0; kt < K; kt += 32) {
    // ---- stage A tile 128x32 -> LDS (bf16) ----
    if constexpr (ABF) {
#if HAVE_TDM
      if (wave == 0) {                          // one wave drives the TDM
        unsigned lds = (unsigned)(size_t)(&As[0][0]);
        unsigned long long ga =
            (unsigned long long)(const void*)(Ab + (size_t)rowBase * K + kt);
        v4u g0; v8i g1;
        v4i g2 = {0, 0, 0, 0}, g3 = {0, 0, 0, 0};
        v8i g4 = {0, 0, 0, 0, 0, 0, 0, 0};
        // D# group0: count=1 | lds_addr | global_addr[56:0] | type=2
        g0[0] = 1u;
        g0[1] = lds;
        g0[2] = (unsigned)(ga & 0xFFFFFFFFu);
        g0[3] = (unsigned)((ga >> 32) & 0x01FFFFFFu) | (2u << 30);
        // D# group1: data_size=2B, pad_enable, pad every 16 DW by 4 DW (stride 40 elems)
        unsigned td0 = (unsigned)K, td1 = (unsigned)M, st0 = (unsigned)K;
        g1[0] = (int)((1u << 16) | (1u << 20) | (3u << 22) | (3u << 25));
        g1[1] = (int)((td0 & 0xFFFFu) << 16);                       // tensor_dim0 lo
        g1[2] = (int)(((td0 >> 16) & 0xFFFFu) | ((td1 & 0xFFFFu) << 16));
        g1[3] = (int)(((td1 >> 16) & 0xFFFFu) | (32u << 16));       // tile_dim0 = 32
        g1[4] = (int)128u;                                          // tile_dim1 = 128
        g1[5] = (int)st0;                                           // dim0 stride lo
        g1[6] = 0;
        g1[7] = 0;
        __builtin_amdgcn_tensor_load_to_lds(g0, g1, g2, g3, g4, 0);
      }
#else
      #pragma unroll
      for (int i = 0; i < 2; ++i) {             // 512 granules of 8 bf16
        int g = tid + i * 256;
        int r = g >> 2, cc = (g & 3) * 8;
        *(uint4*)&As[r][cc] =
            *(const uint4*)(Ab + (size_t)(rowBase + r) * K + kt + cc);
      }
#endif
    } else {
      #pragma unroll
      for (int i = 0; i < 4; ++i) {             // 1024 granules of 4 floats
        int g = tid + i * 256;
        int r = g >> 3, cc = (g & 7) * 4;
        const float4 vv = *(const float4*)(Af + (size_t)(rowBase + r) * K + kt + cc);
        As[r][cc + 0] = tobf(vv.x); As[r][cc + 1] = tobf(vv.y);
        As[r][cc + 2] = tobf(vv.z); As[r][cc + 3] = tobf(vv.w);
      }
    }
    // ---- stage W tile 32x128 transposed -> LDS (bf16) ----
    #pragma unroll
    for (int i = 0; i < 4; ++i) {               // 1024 granules of 4 floats
      int g = tid + i * 256;
      int kk = g >> 5, nn = (g & 31) * 4;
      const float4 wv = *(const float4*)(W + (size_t)(kt + kk) * N + colBase + nn);
      Bs[nn + 0][kk] = tobf(wv.x); Bs[nn + 1][kk] = tobf(wv.y);
      Bs[nn + 2][kk] = tobf(wv.z); Bs[nn + 3][kk] = tobf(wv.w);
    }
    if (kt + 32 < K)                            // pull next W tile toward L2
      __builtin_prefetch(W + (size_t)(kt + 32 + (tid & 31)) * N + colBase + (tid >> 5) * 16, 0, 1);
#if HAVE_TDM
    if (ABF && wave == 0) __builtin_amdgcn_s_wait_tensorcnt(0);
#endif
    __syncthreads();

    v16bf afr[2], bfr[4];
    const int abase = (lane < 16) ? 0 : 16;     // contiguous k = 16h + s
    #pragma unroll
    for (int m = 0; m < 2; ++m) {
      int row = wm * 32 + m * 16 + ln15;
      #pragma unroll
      for (int e = 0; e < 8; ++e) {
        afr[m][e]     = As[row][abase + e];
        afr[m][8 + e] = As[row][abase + 8 + e];
      }
    }
    const int bbase = (lane < 16) ? 0 : 16;     // contiguous k = 16h + s (same as A)
    #pragma unroll
    for (int n = 0; n < 4; ++n) {
      int col = wn * 64 + n * 16 + ln15;
      #pragma unroll
      for (int e = 0; e < 16; ++e) bfr[n][e] = Bs[col][bbase + e];
    }
    #pragma unroll
    for (int m = 0; m < 2; ++m)
      #pragma unroll
      for (int n = 0; n < 4; ++n)
        acc[m][n] = __builtin_amdgcn_wmma_f32_16x16x32_bf16(
            false, afr[m], false, bfr[n], (short)0, acc[m][n], false, false);
    __syncthreads();
  }

  const int hl = lane >> 4;
  #pragma unroll
  for (int m = 0; m < 2; ++m) {
    #pragma unroll
    for (int n = 0; n < 4; ++n) {
      int col = colBase + wn * 64 + n * 16 + ln15;
      float bv = bias ? bias[col] : 0.f;
      #pragma unroll
      for (int r = 0; r < 8; ++r) {
        int row = rowBase + wm * 32 + m * 16 + hl * 8 + r;
        float v = acc[m][n][r] + bv;
        if (ACT == 1) v = v > 0.f ? v : expm1f(v);   // ELU
        size_t off = (size_t)row * N + col;
        if (outF) outF[off] = v;
        if (outB) outB[off] = tobf(v);
      }
    }
  }
}

// ---------------------------------------------------------------------------
// ProbSparse measure: M[b,h,l] = max_j(q.k_samp_j) - sum_j(q.k_samp_j)/L
// one wave per (b,h,l)
// ---------------------------------------------------------------------------
__global__ __launch_bounds__(32)
void sparse_measure(const float* __restrict__ q, const float* __restrict__ k,
                    const int* __restrict__ samp, float* __restrict__ Mout, int u)
{
  int bid = blockIdx.x;                  // b*H*L + h*L + l
  int l = bid % Ln;
  int h = (bid / Ln) % Hn;
  int b = bid / (Ln * Hn);
  int lane = threadIdx.x;
  const float* qrow = q + ((size_t)(b * Ln + l)) * Dn + h * DKn;
  float q0 = qrow[lane], q1 = qrow[32 + lane];
  float mx = -3.0e38f, sm = 0.f;
  for (int j = 0; j < u; ++j) {
    int idx = samp[l * u + j];
    const float* krow = k + ((size_t)(b * Ln + idx)) * Dn + h * DKn;
    float p = q0 * krow[lane] + q1 * krow[32 + lane];
    #pragma unroll
    for (int m2 = 16; m2 >= 1; m2 >>= 1) p += __shfl_xor(p, m2, 32);
    mx = fmaxf(mx, p);
    sm += p;
  }
  if (lane == 0) Mout[bid] = mx - sm / (float)Ln;
}

// top-u per (b,h): iterative block argmax (ties -> lower index, matches top_k)
__global__ __launch_bounds__(256)
void topk_kernel(const float* __restrict__ Mv, int* __restrict__ top, int u)
{
  __shared__ float vals[Ln];
  __shared__ float rv[256];
  __shared__ int   ri[256];
  int bh = blockIdx.x, t = threadIdx.x;
  for (int i = t; i < Ln; i += 256) vals[i] = Mv[(size_t)bh * Ln + i];
  __syncthreads();
  for (int it = 0; it < u; ++it) {
    float bv = -3.0e38f; int bi = 0x7fffffff;
    for (int i = t; i < Ln; i += 256) {
      float v = vals[i];
      if (v > bv || (v == bv && i < bi)) { bv = v; bi = i; }
    }
    rv[t] = bv; ri[t] = bi;
    __syncthreads();
    for (int s2 = 128; s2 >= 1; s2 >>= 1) {
      if (t < s2) {
        float ov = rv[t + s2]; int oi = ri[t + s2];
        if (ov > rv[t] || (ov == rv[t] && oi < ri[t])) { rv[t] = ov; ri[t] = oi; }
      }
      __syncthreads();
    }
    if (t == 0) { top[bh * u + it] = ri[0]; vals[ri[0]] = -3.0e38f; }
    __syncthreads();
  }
}

// attention for the u selected query rows; one block per (b,h,i)
__global__ __launch_bounds__(256)
void sparse_attn(const float* __restrict__ q, const float* __restrict__ k,
                 const float* __restrict__ v, const int* __restrict__ top,
                 float* __restrict__ outtop, int u)
{
  __shared__ float qv[DKn];
  __shared__ float sc[Ln];
  __shared__ float red[256];
  int bh = blockIdx.y, i = blockIdx.x, t = threadIdx.x;
  int b = bh / Hn, h = bh % Hn;
  int lq = top[bh * u + i];
  if (t < DKn) qv[t] = q[((size_t)(b * Ln + lq)) * Dn + h * DKn + t];
  __syncthreads();
  float lmax = -3.0e38f;
  for (int l = t; l < Ln; l += 256) {
    const float* krow = k + ((size_t)(b * Ln + l)) * Dn + h * DKn;
    float d = 0.f;
    #pragma unroll
    for (int e = 0; e < DKn; ++e) d += qv[e] * krow[e];
    d *= 0.125f;                                  // 1/sqrt(DK)
    sc[l] = d;
    lmax = fmaxf(lmax, d);
  }
  red[t] = lmax; __syncthreads();
  for (int s2 = 128; s2 >= 1; s2 >>= 1) { if (t < s2) red[t] = fmaxf(red[t], red[t + s2]); __syncthreads(); }
  float mx = red[0];
  __syncthreads();
  float lsum = 0.f;
  for (int l = t; l < Ln; l += 256) { float p = __expf(sc[l] - mx); sc[l] = p; lsum += p; }
  red[t] = lsum; __syncthreads();
  for (int s2 = 128; s2 >= 1; s2 >>= 1) { if (t < s2) red[t] += red[t + s2]; __syncthreads(); }
  float inv = 1.f / red[0];
  __syncthreads();
  int d0 = t & 63, g = t >> 6;
  float acc = 0.f;
  for (int l = g * 256; l < (g + 1) * 256; ++l)
    acc += sc[l] * v[((size_t)(b * Ln + l)) * Dn + h * DVn + d0];
  red[t] = acc; __syncthreads();
  if (g == 0) {
    float s = red[d0] + red[64 + d0] + red[128 + d0] + red[192 + d0];
    outtop[((size_t)bh * u + i) * DVn + d0] = s * inv;
  }
}

__global__ __launch_bounds__(64)
void vmean_kernel(const float* __restrict__ v, float* __restrict__ vm)
{
  int bh = blockIdx.x, d = threadIdx.x;
  int b = bh / Hn, h = bh % Hn;
  float s = 0.f;
  for (int l = 0; l < Ln; ++l) s += v[((size_t)(b * Ln + l)) * Dn + h * DVn + d];
  vm[bh * DVn + d] = s / (float)Ln;
}

__global__ __launch_bounds__(256)
void fill_ctx(const float* __restrict__ vm, bf16_t* __restrict__ O)
{
  size_t idx = (size_t)blockIdx.x * 256 + threadIdx.x;   // < B*L*512
  int c = (int)(idx % Dn);
  size_t bl = idx / Dn;
  int b = (int)(bl / Ln);
  int h = c / DVn, d = c % DVn;
  O[idx] = tobf(vm[(b * Hn + h) * DVn + d]);
}

__global__ __launch_bounds__(64)
void scatter_top(const float* __restrict__ outtop, const int* __restrict__ top,
                 bf16_t* __restrict__ O, int u)
{
  int gi = blockIdx.x;                   // bh*u + i
  int bh = gi / u, i = gi % u;
  int b = bh / Hn, h = bh % Hn;
  int t = threadIdx.x;
  int lq = top[bh * u + i];
  O[((size_t)(b * Ln + lq)) * Dn + h * DVn + t] = tobf(outtop[((size_t)bh * u + i) * DVn + t]);
}

// y = LN(x + a) * g + b ; emits fp32 (residual chain) and optional bf16 copy
__global__ __launch_bounds__(256)
void resid_ln(const float* __restrict__ x, const float* __restrict__ a,
              const float* __restrict__ g, const float* __restrict__ bb,
              float* __restrict__ outF, bf16_t* __restrict__ outB)
{
  __shared__ float s1[256], s2[256];
  int row = blockIdx.x, t = threadIdx.x;
  const float* xr = x + (size_t)row * Dn;
  const float* ar = a + (size_t)row * Dn;
  float v0 = xr[t] + ar[t];
  float v1 = xr[t + 256] + ar[t + 256];
  s1[t] = v0 + v1;
  s2[t] = v0 * v0 + v1 * v1;
  __syncthreads();
  for (int s = 128; s >= 1; s >>= 1) { if (t < s) { s1[t] += s1[t + s]; s2[t] += s2[t + s]; } __syncthreads(); }
  float mean = s1[0] / (float)Dn;
  float var  = s2[0] / (float)Dn - mean * mean;
  float inv  = rsqrtf(var + 0.001f);
  float y0 = g[t] * (v0 - mean) * inv + bb[t];
  float y1 = g[t + 256] * (v1 - mean) * inv + bb[t + 256];
  size_t off = (size_t)row * Dn;
  if (outF) { outF[off + t] = y0; outF[off + t + 256] = y1; }
  if (outB) { outB[off + t] = tobf(y0); outB[off + t + 256] = tobf(y1); }
}

// ---------------------------------------------------------------------------
// Flash causal cross-attention, bf16 WMMA, fp32 online softmax.
// One wave per (b, h, 16-row q tile); S streamed in 32-wide windows.
// ---------------------------------------------------------------------------
__global__ __launch_bounds__(32)
void flash_cross(const bf16_t* __restrict__ Q, const bf16_t* __restrict__ Kb,
                 const bf16_t* __restrict__ Vb, bf16_t* __restrict__ O)
{
  __shared__ __align__(16) bf16_t Vs[32][64];
  __shared__ bf16_t Ps[16][32];
  const int lane = threadIdx.x;
  const int l0 = blockIdx.x * 16;
  const int bh = blockIdx.y;
  const int b = bh / Hn, h = bh % Hn;
  const int hl = lane >> 4, ln15 = lane & 15;

  v16bf aq[2];                                     // Q 16x64 -> two 16x32 A frags
  {
    const bf16_t* qrow = Q + ((size_t)(b * Ln + l0 + ln15)) * (Hn * DKn) + h * DKn;
    const int base = (lane < 16) ? 0 : 16;
    #pragma unroll
    for (int f = 0; f < 2; ++f)
      #pragma unroll
      for (int e = 0; e < 8; ++e) {
        aq[f][e]     = qrow[f * 32 + base + e];
        aq[f][8 + e] = qrow[f * 32 + base + 8 + e];
      }
  }

  v8f oc[4] = {};
  float mrow[8], lrow[8];
  #pragma unroll
  for (int r = 0; r < 8; ++r) { mrow[r] = -3.0e38f; lrow[r] = 0.f; }

  const int nwin = (l0 + 16 + 31) / 32;            // causal: s <= l only
  for (int w = 0; w < nwin; ++w) {
    const int s0 = w * 32;
    {                                              // stage V 32x64 tile
      const bf16_t* vrow = Vb + ((size_t)(b * Sn + s0 + lane)) * (Hn * DVn) + h * DVn;
      const uint4* src = (const uint4*)vrow;
      uint4* dst = (uint4*)(&Vs[lane][0]);
      #pragma unroll
      for (int i = 0; i < 8; ++i) dst[i] = src[i];
    }
    v8f sc[2] = {};                                // two 16x16 score tiles
    {
      const int bbase = (lane < 16) ? 0 : 16;
      #pragma unroll
      for (int t = 0; t < 2; ++t) {
        int s = s0 + t * 16 + ln15;
        const bf16_t* krow = Kb + ((size_t)(b * Sn + s)) * (Hn * DKn) + h * DKn;
        #pragma unroll
        for (int f = 0; f < 2; ++f) {
          v16bf bk;
          #pragma unroll
          for (int e = 0; e < 16; ++e) bk[e] = krow[f * 32 + bbase + e];
          sc[t] = __builtin_amdgcn_wmma_f32_16x16x32_bf16(
              false, aq[f], false, bk, (short)0, sc[t], false, false);
        }
      }
    }
    float alpha[8];                                // online softmax per row
    #pragma unroll
    for (int r = 0; r < 8; ++r) {
      int lq = l0 + hl * 8 + r;
      float v0 = sc[0][r] * 0.125f;
      float v1 = sc[1][r] * 0.125f;
      if (s0 + ln15 > lq)      v0 = -1.0e9f;
      if (s0 + 16 + ln15 > lq) v1 = -1.0e9f;
      float mx = fmaxf(v0, v1);
      #pragma unroll
      for (int m2 = 8; m2 >= 1; m2 >>= 1) mx = fmaxf(mx, __shfl_xor(mx, m2, 32));
      float mn = fmaxf(mrow[r], mx);
      float al = __expf(mrow[r] - mn);
      float p0 = __expf(v0 - mn);
      float p1 = __expf(v1 - mn);
      float ps = p0 + p1;
      #pragma unroll
      for (int m2 = 8; m2 >= 1; m2 >>= 1) ps += __shfl_xor(ps, m2, 32);
      lrow[r] = lrow[r] * al + ps;
      mrow[r] = mn;
      alpha[r] = al;
      Ps[hl * 8 + r][ln15]      = tobf(p0);        // P -> LDS (C-layout write)
      Ps[hl * 8 + r][16 + ln15] = tobf(p1);
    }
    #pragma unroll
    for (int t = 0; t < 4; ++t)
      #pragma unroll
      for (int r = 0; r < 8; ++r) oc[t][r] *= alpha[r];
    __syncthreads();
    v16bf ap;                                      // P back as A fragment
    {
      const int abase = (lane < 16) ? 0 : 16;
      #pragma unroll
      for (int e = 0; e < 8; ++e) {
        ap[e]     = Ps[ln15][abase + e];
        ap[8 + e] = Ps[ln15][abase + 8 + e];
      }
    }
    const int bb2 = (lane < 16) ? 0 : 16;
    #pragma unroll
    for (int t = 0; t < 4; ++t) {                  // P(16x32) @ V(32x16) x4
      v16bf bv;
      #pragma unroll
      for (int e = 0; e < 16; ++e) bv[e] = Vs[bb2 + e][t * 16 + ln15];
      oc[t] = __builtin_amdgcn_wmma_f32_16x16x32_bf16(
          false, ap, false, bv, (short)0, oc[t], false, false);
    }
    __syncthreads();
  }
  #pragma unroll
  for (int t = 0; t < 4; ++t)
    #pragma unroll
    for (int r = 0; r < 8; ++r) {
      int row = l0 + hl * 8 + r;
      float v = oc[t][r] / lrow[r];
      O[((size_t)(b * Ln + row)) * (Hn * DVn) + h * DVn + t * 16 + ln15] = tobf(v);
    }
}

// ---------------------------------------------------------------------------
extern "C" void kernel_launch(void* const* d_in, const int* in_sizes, int n_in,
                              void* d_out, int out_size, void* d_ws, size_t ws_size,
                              hipStream_t stream)
{
  (void)n_in; (void)out_size; (void)ws_size;
  const float* x       = (const float*)d_in[0];
  const float* context = (const float*)d_in[1];
  const int*   samp    = (const int*)d_in[2];
  const float* ps[16]; const float* ca[16];
  for (int i = 0; i < 16; ++i) { ps[i] = (const float*)d_in[3 + i]; ca[i] = (const float*)d_in[19 + i]; }
  enum { WQ = 0, BQ, WK, BKb, WV, BV, WO, BO, LN1G, LN1B, W1, B1, W2, B2, LN2G, LN2B };
  const int u = in_sizes[2] / Ln;   // 28

  const size_t MB = 1024 * 1024;
  char* wsb = (char*)d_ws;
  float*  qf  = (float*)(wsb + 0 * MB);     // 16MB  (later q2b region)
  float*  kf  = (float*)(wsb + 16 * MB);    // 16MB  (later k2b region)
  float*  vf  = (float*)(wsb + 32 * MB);    // 16MB  (later v2b region)
  bf16_t* q2b = (bf16_t*)(wsb + 0 * MB);    // 8MB
  bf16_t* k2b = (bf16_t*)(wsb + 16 * MB);   // 16MB
  bf16_t* v2b = (bf16_t*)(wsb + 32 * MB);   // 16MB
  float*  gf  = (float*)(wsb + 48 * MB);    // 16MB  generic fp32 GEMM out
  float*  xf0 = (float*)(wsb + 64 * MB);    // 16MB
  float*  xf1 = (float*)(wsb + 80 * MB);    // 16MB
  bf16_t* xb  = (bf16_t*)(wsb + 96 * MB);   // 8MB
  bf16_t* hb  = (bf16_t*)(wsb + 104 * MB);  // 32MB  FFN hidden (bf16)
  bf16_t* ob  = (bf16_t*)(wsb + 136 * MB);  // 8MB   attention output (bf16)
  float*  Mm  = (float*)(wsb + 144 * MB);             // 256KB
  int*    top = (int*)  (wsb + 144 * MB + 256 * 1024);
  float*  ot  = (float*)(wsb + 144 * MB + 512 * 1024);// u*64*64*4
  float*  vm  = (float*)(wsb + 145 * MB);             // 16KB

  const int ML = Bn * Ln, MS = Bn * Sn;
  dim3 b256(256), b32(32), b64(64);

  // ---- ProbSparse self-attention ----
  gemm_wmma<0, false><<<dim3(Dn / 128, ML / 128), b256, 0, stream>>>(x, ps[WQ], ps[BQ], qf, nullptr, ML, Dn, Dn);
  gemm_wmma<0, false><<<dim3(Dn / 128, ML / 128), b256, 0, stream>>>(x, ps[WK], ps[BKb], kf, nullptr, ML, Dn, Dn);
  gemm_wmma<0, false><<<dim3(Dn / 128, ML / 128), b256, 0, stream>>>(x, ps[WV], ps[BV], vf, nullptr, ML, Dn, Dn);
  sparse_measure<<<Bn * Hn * Ln, b32, 0, stream>>>(qf, kf, samp, Mm, u);
  topk_kernel<<<Bn * Hn, b256, 0, stream>>>(Mm, top, u);
  sparse_attn<<<dim3(u, Bn * Hn), b256, 0, stream>>>(qf, kf, vf, top, ot, u);
  vmean_kernel<<<Bn * Hn, b64, 0, stream>>>(vf, vm);
  fill_ctx<<<(ML * Dn) / 256, b256, 0, stream>>>(vm, ob);
  scatter_top<<<Bn * Hn * u, b64, 0, stream>>>(ot, top, ob, u);
  gemm_wmma<0, true><<<dim3(Dn / 128, ML / 128), b256, 0, stream>>>(ob, ps[WO], ps[BO], gf, nullptr, ML, Dn, Dn);
  resid_ln<<<ML, b256, 0, stream>>>(x, gf, ps[LN1G], ps[LN1B], xf0, xb);
  // FFN 1
  gemm_wmma<1, true><<<dim3(HIDn / 128, ML / 128), b256, 0, stream>>>(xb, ps[W1], ps[B1], nullptr, hb, ML, HIDn, Dn);
  gemm_wmma<0, true><<<dim3(Dn / 128, ML / 128), b256, 0, stream>>>(hb, ps[W2], ps[B2], gf, nullptr, ML, Dn, HIDn);
  resid_ln<<<ML, b256, 0, stream>>>(xf0, gf, ps[LN2G], ps[LN2B], xf1, xb);

  // ---- Causal cross-attention ----
  gemm_wmma<0, true ><<<dim3(Dn / 128, ML / 128), b256, 0, stream>>>(xb, ca[WQ], ca[BQ], nullptr, q2b, ML, Dn, Dn);
  gemm_wmma<0, false><<<dim3(Dn / 128, MS / 128), b256, 0, stream>>>(context, ca[WK], ca[BKb], nullptr, k2b, MS, Dn, Dn);
  gemm_wmma<0, false><<<dim3(Dn / 128, MS / 128), b256, 0, stream>>>(context, ca[WV], ca[BV], nullptr, v2b, MS, Dn, Dn);
  flash_cross<<<dim3(Ln / 16, Bn * Hn), b32, 0, stream>>>(q2b, k2b, v2b, ob);
  gemm_wmma<0, true><<<dim3(Dn / 128, ML / 128), b256, 0, stream>>>(ob, ca[WO], ca[BO], gf, nullptr, ML, Dn, Dn);
  resid_ln<<<ML, b256, 0, stream>>>(xf1, gf, ca[LN1G], ca[LN1B], xf0, xb);
  // FFN 2
  gemm_wmma<1, true><<<dim3(HIDn / 128, ML / 128), b256, 0, stream>>>(xb, ca[W1], ca[B1], nullptr, hb, ML, HIDn, Dn);
  gemm_wmma<0, true><<<dim3(Dn / 128, ML / 128), b256, 0, stream>>>(hb, ca[W2], ca[B2], gf, nullptr, ML, Dn, HIDn);
  resid_ln<<<ML, b256, 0, stream>>>(xf0, gf, ca[LN2G], ca[LN2B], (float*)d_out, nullptr);
}
